// BahdanauAttention_16501264351731
// MI455X (gfx1250) — compile-verified, matrix operand order checked by reading, not run on hardware
//
#include <hip/hip_runtime.h>
#include <math.h>

#define B_  32
#define S_  2048
#define H_  1024
#define HL_ 2048

typedef __attribute__((ext_vector_type(16))) __bf16 v16bf;
typedef __attribute__((ext_vector_type(8)))  float  v8f;

__device__ __forceinline__ unsigned pack2bf16(float x, float y) {
  union { float f; unsigned u; } ax, ay;
  ax.f = x; ay.f = y;
  unsigned rx = (ax.u + 0x7FFFu + ((ax.u >> 16) & 1u)) >> 16;  // RNE
  unsigned ry = (ay.u + 0x7FFFu + ((ay.u >> 16) & 1u)) >> 16;
  return rx | (ry << 16);
}

// Branch-free tanh: (e^{2x}-1)/(e^{2x}+1), clamped so e^{2x} stays finite.
__device__ __forceinline__ float tanh_fast(float x) {
  x = fminf(fmaxf(x, -15.0f), 15.0f);
  const float t = __expf(2.0f * x);
  return (t - 1.0f) * __builtin_amdgcn_rcpf(t + 1.0f);
}

// ---------------------------------------------------------------------------
// K1: qq[b,h] = query[b,:] @ Wa_w[:,h] + Wa_b[h] + Ua_b[h]
// grid (4, 32), block 256
// ---------------------------------------------------------------------------
__global__ void __launch_bounds__(256)
qproj_kernel(const float* __restrict__ query, const float* __restrict__ Wa_w,
             const float* __restrict__ Wa_b,  const float* __restrict__ Ua_b,
             float* __restrict__ qq) {
  const int h = blockIdx.x * 256 + threadIdx.x;   // 0..1023
  const int b = blockIdx.y;
  const float* qrow = query + (size_t)b * HL_;
  float acc = Wa_b[h] + Ua_b[h];
  for (int d = 0; d < HL_; ++d)
    acc = fmaf(qrow[d], Wa_w[(size_t)d * H_ + h], acc);
  qq[b * H_ + h] = acc;
}

// ---------------------------------------------------------------------------
// K2: Ua_w f32 [K=1024][N=1024] -> bf16 in WMMA-B operand layout.
// Tile = 16(n) x 32(k). tile index = nt*32 + kt. Within a tile:
//   lane = (n%16) | (((kk>>3)&1) << 4), slot e = ((kk>=16)?8:0) + (kk&7)
// Per-lane 16 bf16 stored contiguously -> GEMM loads one v16bf per lane.
// grid (256), block 256 (8 tiles/block, 1 wave per tile)
// ---------------------------------------------------------------------------
__global__ void __launch_bounds__(256)
swizzleB_kernel(const float* __restrict__ Ua_w, unsigned short* __restrict__ Ub) {
  const int tid  = threadIdx.x;
  const int tile = blockIdx.x * 8 + (tid >> 5);   // 0..2047
  const int lane = tid & 31;
  const int nt = tile >> 5;                       // 0..63
  const int kt = tile & 31;                       // 0..31
  const int n  = (nt << 4) + (lane & 15);
  const int hi = (lane >> 4) << 3;
  unsigned* outp = reinterpret_cast<unsigned*>(Ub + (size_t)tile * 512 + lane * 16);
#pragma unroll
  for (int j = 0; j < 8; ++j) {
    const int e0  = 2 * j;
    const int kk0 = ((e0 >= 8) ? 16 : 0) + (e0 & 7) + hi;     // kk1 = kk0+1
    const float x = Ua_w[(size_t)(kt * 32 + kk0)     * H_ + n];
    const float y = Ua_w[(size_t)(kt * 32 + kk0 + 1) * H_ + n];
    outp[j] = pack2bf16(x, y);
  }
}

// ---------------------------------------------------------------------------
// K3: fused  scores[b,s] = tanh(keys[b,s,:]@Ua_w + qq[b,:]) @ Va_w + Va_b
// One block = one batch x 32 rows. 256 threads = 8 waves.
// keys slab staged f32->bf16 into LDS in WMMA-A layout; Ua streamed from the
// pre-swizzled bf16 buffer (L2 resident). Each wave drives TWO independent
// 16x16x32 bf16 WMMA accumulator chains sharing one A tile.
// grid (64, 32), block 256
// ---------------------------------------------------------------------------
__global__ void __launch_bounds__(256)
attn_gemm_kernel(const float* __restrict__ keys,
                 const unsigned short* __restrict__ Ub,
                 const float* __restrict__ qq,
                 const float* __restrict__ Va_w,
                 const float* __restrict__ Va_b,
                 float* __restrict__ scores) {
  __shared__ __align__(32) unsigned short ldsA[64 * 512];  // 64 KB: 2 mt x 32 kt tiles
  __shared__ float ldsQQ[H_];
  __shared__ float ldsVa[H_];
  __shared__ float scoreRed[32];

  const int tid = threadIdx.x;
  const int b   = blockIdx.y;
  const int m0  = blockIdx.x * 32;

  for (int i = tid; i < H_; i += 256) {
    ldsQQ[i] = qq[b * H_ + i];
    ldsVa[i] = Va_w[i];
  }
  if (tid < 32) scoreRed[tid] = 0.0f;

  // Stage A: keys[b, m0:m0+32, 0:1024] -> bf16 WMMA-A layout in LDS.
  // Coalesced float2 global reads, packed 2xbf16 b32 LDS stores.
  const float* krows = keys + ((size_t)b * S_ + m0) * H_;
  for (int it = 0; it < 64; ++it) {
    const int p = it * 256 + tid;       // pair index [0,16384)
    const int r = p >> 9;               // row 0..31
    const int k = (p & 511) << 1;       // even col 0..1022
    const float2 v = *reinterpret_cast<const float2*>(krows + (size_t)r * H_ + k);
    const unsigned packed = pack2bf16(v.x, v.y);
    const int kt   = k >> 5, kk = k & 31;
    const int e    = ((kk >> 4) << 3) | (kk & 7);
    const int lane = (r & 15) | (((kk >> 3) & 1) << 4);
    const int idx  = (((r >> 4) << 5) | kt) * 512 + lane * 16 + e;
    *reinterpret_cast<unsigned*>(&ldsA[idx]) = packed;
  }
  __syncthreads();

  const int w     = tid >> 5;
  const int lane  = tid & 31;
  const int mt    = w & 1;            // which 16-row tile
  const int grp   = w >> 1;           // 0..3: n-tile pair group
  const int nlo   = lane & 15;
  const int mbase = (mt << 4) + ((lane >> 4) << 3);

  float part[8];
#pragma unroll
  for (int j = 0; j < 8; ++j) part[j] = 0.0f;

  const unsigned short* ap0 = &ldsA[(mt << 5) * 512 + lane * 16];

  for (int nti = 0; nti < 8; ++nti) {
    const int nt0 = (grp << 1) + (nti << 3);      // {0,2,4,6} + 8*i
    v8f acc0 = {};
    v8f acc1 = {};
    const unsigned short* bp0 = Ub + (size_t)nt0 * (32 * 512) + lane * 16;
    const unsigned short* bp1 = bp0 + 32 * 512;   // nt0 + 1
#pragma unroll 2
    for (int kt = 0; kt < 32; ++kt) {
      v16bf amat = *reinterpret_cast<const v16bf*>(ap0 + kt * 512);
      v16bf b0   = *reinterpret_cast<const v16bf*>(bp0 + kt * 512);
      v16bf b1   = *reinterpret_cast<const v16bf*>(bp1 + kt * 512);
      acc0 = __builtin_amdgcn_wmma_f32_16x16x32_bf16(
          false, amat, false, b0, (short)0, acc0, false, false);
      acc1 = __builtin_amdgcn_wmma_f32_16x16x32_bf16(
          false, amat, false, b1, (short)0, acc1, false, false);
    }
    // Fused epilogue: e = tanh(acc + q[n]); partial score += e * Va[n]
    const int   n0  = (nt0 << 4) + nlo;
    const int   n1  = n0 + 16;
    const float qn0 = ldsQQ[n0], va0 = ldsVa[n0];
    const float qn1 = ldsQQ[n1], va1 = ldsVa[n1];
#pragma unroll
    for (int j = 0; j < 8; ++j) {
      part[j] += tanh_fast(acc0[j] + qn0) * va0;
      part[j] += tanh_fast(acc1[j] + qn1) * va1;
    }
  }
  // Reduce across lanes (16 lanes share each row) with LDS float atomics.
#pragma unroll
  for (int j = 0; j < 8; ++j)
    atomicAdd(&scoreRed[mbase + j], part[j]);
  __syncthreads();

  if (tid < 32)
    scores[b * S_ + m0 + tid] = scoreRed[tid] + Va_b[0];
}

// ---------------------------------------------------------------------------
// K4: softmax over S per batch. grid (32), block 256
// ---------------------------------------------------------------------------
__global__ void __launch_bounds__(256)
softmax_kernel(const float* __restrict__ scores, float* __restrict__ weights) {
  __shared__ float red[256];
  const int b = blockIdx.x, tid = threadIdx.x;
  const float* row = scores + b * S_;

  float m = -1e30f;
  for (int s = tid; s < S_; s += 256) m = fmaxf(m, row[s]);
  red[tid] = m; __syncthreads();
  for (int o = 128; o > 0; o >>= 1) {
    if (tid < o) red[tid] = fmaxf(red[tid], red[tid + o]);
    __syncthreads();
  }
  m = red[0]; __syncthreads();

  float sum = 0.0f;
  for (int s = tid; s < S_; s += 256) sum += __expf(row[s] - m);
  red[tid] = sum; __syncthreads();
  for (int o = 128; o > 0; o >>= 1) {
    if (tid < o) red[tid] += red[tid + o];
    __syncthreads();
  }
  const float inv = __builtin_amdgcn_rcpf(red[0]);
  for (int s = tid; s < S_; s += 256)
    weights[b * S_ + s] = __expf(row[s] - m) * inv;
}

// ---------------------------------------------------------------------------
// K5: context[b,h] = sum_s weights[b,s] * keys[b,s,h]. grid (4, 32), block 256
// ---------------------------------------------------------------------------
__global__ void __launch_bounds__(256)
context_kernel(const float* __restrict__ keys, const float* __restrict__ weights,
               float* __restrict__ ctx) {
  const int h = blockIdx.x * 256 + threadIdx.x;
  const int b = blockIdx.y;
  const float* kb = keys + (size_t)b * S_ * H_;
  const float* wb = weights + b * S_;
  float acc = 0.0f;
  for (int s = 0; s < S_; ++s)
    acc = fmaf(wb[s], kb[(size_t)s * H_ + h], acc);
  ctx[b * H_ + h] = acc;
}

// ---------------------------------------------------------------------------
extern "C" void kernel_launch(void* const* d_in, const int* in_sizes, int n_in,
                              void* d_out, int out_size, void* d_ws, size_t ws_size,
                              hipStream_t stream) {
  const float* query = (const float*)d_in[0];
  const float* keys  = (const float*)d_in[1];
  const float* Wa_w  = (const float*)d_in[2];
  const float* Wa_b  = (const float*)d_in[3];
  const float* Ua_w  = (const float*)d_in[4];
  const float* Ua_b  = (const float*)d_in[5];
  const float* Va_w  = (const float*)d_in[6];
  const float* Va_b  = (const float*)d_in[7];

  float* out     = (float*)d_out;
  float* ctx     = out;                 // [32,1,1024]
  float* weights = out + B_ * H_;       // [32,1,2048]

  char* ws = (char*)d_ws;
  float*          qqbuf  = (float*)ws;                                  // 128 KB
  unsigned short* Ubbuf  = (unsigned short*)(ws + (128 << 10));         // 2 MB bf16
  float*          scores = (float*)(ws + (128 << 10) + (2 << 20));      // 256 KB

  qproj_kernel   <<<dim3(4, 32),  256, 0, stream>>>(query, Wa_w, Wa_b, Ua_b, qqbuf);
  swizzleB_kernel<<<dim3(256),    256, 0, stream>>>(Ua_w, Ubbuf);
  attn_gemm_kernel<<<dim3(64, 32),256, 0, stream>>>(keys, Ubbuf, qqbuf, Va_w, Va_b, scores);
  softmax_kernel <<<dim3(32),     256, 0, stream>>>(scores, weights);
  context_kernel <<<dim3(4, 32),  256, 0, stream>>>(keys, weights, ctx);
}
